// metaDynamicGCN_11897059410449
// MI455X (gfx1250) — compile-verified, matrix operand order checked by reading, not run on hardware
//
#include <hip/hip_runtime.h>

typedef __attribute__((ext_vector_type(16))) _Float16 v16h;
typedef __attribute__((ext_vector_type(8)))  float    v8f;
typedef _Float16 half_t;

// ---------------------------------------------------------------------------
// Kernel 1: fold the two diffusion weight matrices (hop-0 rows only, i<64),
// convert to f16, and store pre-swizzled into WMMA B-fragment layout so the
// main kernel loads each B fragment with ONE 32B vector load.
// ---------------------------------------------------------------------------
__global__ void prep_weights(const float* __restrict__ Wz,
                             const float* __restrict__ Wh,
                             half_t* __restrict__ wsZ,
                             half_t* __restrict__ wsH) {
  const float* src = (blockIdx.x == 0) ? Wz : Wh;   // shape (2,1,128,64)
  half_t*      dst = (blockIdx.x == 0) ? wsZ : wsH;
  const int t    = threadIdx.x;      // 0..255
  const int frag = t >> 5;           // 0..7
  const int lane = t & 31;
  const int c    = frag >> 1;        // output-column chunk 0..3
  const int s    = frag & 1;         // K step 0..1
  const int j    = c * 16 + (lane & 15);
  const int kb   = s * 32 + ((lane & 16) ? 8 : 0);
  half_t* o = dst + ((size_t)frag * 32 + lane) * 16;
#pragma unroll
  for (int e = 0; e < 16; ++e) {
    const int K = (e < 8) ? (kb + e) : (kb + 16 + (e - 8));
    // W[0,0,K,j] + W[1,0,K,j]; group stride = 128*64
    const float v = src[K * 64 + j] + src[128 * 64 + K * 64 + j];
    o[e] = (half_t)v;
  }
}

// Fast activations built on raw CDNA5 trans ops (v_exp_f32 = 2^x, v_rcp_f32).
// one_minus_sigmoid(a) = 1/(1+e^a); tanh(b) = 1 - 2/(e^{2b}+1).
// Both saturate correctly through inf -> rcp -> 0 at the extremes.
#define L2E 1.4426950408889634f
__device__ __forceinline__ float one_minus_sigmoid(float a) {
  return __builtin_amdgcn_rcpf(1.0f + __builtin_amdgcn_exp2f(a * L2E));
}
__device__ __forceinline__ float fast_tanh(float b) {
  const float r = __builtin_amdgcn_rcpf(
      __builtin_amdgcn_exp2f(b * (2.0f * L2E)) + 1.0f);
  return __builtin_fmaf(-2.0f, r, 1.0f);
}

// ---------------------------------------------------------------------------
// Kernel 2: fused sigmoid/tanh GRU (H0==0) + relu + linear head.
// One wave handles a 16-node tile: 16 v_wmma_f32_16x16x32_f16 per tile,
// elementwise + the 64-wide reduction done in-register with shfl_xor.
// ---------------------------------------------------------------------------
__global__ __launch_bounds__(256) void dcrnn_fused(
    const float* __restrict__ x,
    const float* __restrict__ bz,
    const float* __restrict__ bh,
    const float* __restrict__ wlin,
    const float* __restrict__ blin,
    const half_t* __restrict__ wsZ,
    const half_t* __restrict__ wsH,
    float* __restrict__ out,
    int n) {
  const int lane = threadIdx.x & 31;
  const int wave = threadIdx.x >> 5;
  const int tile = blockIdx.x * 8 + wave;
  const int numTiles = (n + 15) >> 4;
  if (tile >= numTiles) return;      // wave-uniform: EXEC stays all-ones

  const int tileBase = tile << 4;
  const int l16 = lane & 15;
  int row = tileBase + l16;
  if (row >= n) row = n - 1;         // clamp (not mask) to keep EXEC full

  // ---- A fragments: x rows, f32 -> f16, CDNA5 16-bit A layout ------------
  const float* xr   = x + (size_t)row * 64;
  const int    koff = (lane & 16) ? 8 : 0;
  v16h a[2];
#pragma unroll
  for (int s = 0; s < 2; ++s) {
    const int kb = s * 32 + koff;
    const float4 f0 = *(const float4*)(xr + kb);
    const float4 f1 = *(const float4*)(xr + kb + 4);
    const float4 f2 = *(const float4*)(xr + kb + 16);
    const float4 f3 = *(const float4*)(xr + kb + 20);
    v16h av;
    av[0]  = (half_t)f0.x; av[1]  = (half_t)f0.y; av[2]  = (half_t)f0.z; av[3]  = (half_t)f0.w;
    av[4]  = (half_t)f1.x; av[5]  = (half_t)f1.y; av[6]  = (half_t)f1.z; av[7]  = (half_t)f1.w;
    av[8]  = (half_t)f2.x; av[9]  = (half_t)f2.y; av[10] = (half_t)f2.z; av[11] = (half_t)f2.w;
    av[12] = (half_t)f3.x; av[13] = (half_t)f3.y; av[14] = (half_t)f3.z; av[15] = (half_t)f3.w;
    a[s] = av;
  }

  // ---- GEMMs: K=64 as two 16x16x32 WMMAs per 16-col chunk, both gates ----
  v8f accZ[4], accH[4];
#pragma unroll
  for (int c = 0; c < 4; ++c) {
    v8f az = {}; v8f ah = {};
#pragma unroll
    for (int s = 0; s < 2; ++s) {
      const size_t fo = ((size_t)(c * 2 + s) * 32 + lane) * 16;
      const v16h bzf = *(const v16h*)(wsZ + fo);
      az = __builtin_amdgcn_wmma_f32_16x16x32_f16(false, a[s], false, bzf,
                                                  (short)0, az, false, false);
      const v16h bhf = *(const v16h*)(wsH + fo);
      ah = __builtin_amdgcn_wmma_f32_16x16x32_f16(false, a[s], false, bhf,
                                                  (short)0, ah, false, false);
    }
    accZ[c] = az; accH[c] = ah;
  }

  // ---- fused elementwise + linear head (branch-free fast trans ops) ------
  const float bl = blin[0];
  float p[8];
#pragma unroll
  for (int r = 0; r < 8; ++r) p[r] = 0.0f;

#pragma unroll
  for (int c = 0; c < 4; ++c) {
    const int col   = c * 16 + l16;
    const float bzc = bz[col];
    const float bhc = bh[col];
    const float wl  = wlin[col];
#pragma unroll
    for (int r = 0; r < 8; ++r) {
      const float w  = one_minus_sigmoid(accZ[c][r] + bzc);  // 1 - Z
      const float ht = fast_tanh(accH[c][r] + bhc);          // H_tilde
      float hv = w * ht;                                     // H (H0 == 0)
      hv = fmaxf(hv, 0.0f);                                  // relu
      p[r] = __builtin_fmaf(hv, wl, p[r]);                   // h . W_lin
    }
  }

  // reduce across the 16 lanes of each half-wave (C-layout M split)
#pragma unroll
  for (int m = 1; m < 16; m <<= 1) {
#pragma unroll
    for (int r = 0; r < 8; ++r) p[r] += __shfl_xor(p[r], m, 32);
  }

  // lanes 0-15 hold rows 0..7 of the tile, lanes 16-31 hold rows 8..15
  if (lane == 0 || lane == 16) {
    const int rbase = tileBase + ((lane >> 4) ? 8 : 0);
#pragma unroll
    for (int r = 0; r < 8; ++r) {
      const int rr = rbase + r;
      if (rr < n) out[rr] = p[r] + bl;
    }
  }
}

extern "C" void kernel_launch(void* const* d_in, const int* in_sizes, int n_in,
                              void* d_out, int out_size, void* d_ws, size_t ws_size,
                              hipStream_t stream) {
  // setup_inputs order:
  // 0:x 1:edge_index 2:edge_weight 3:Wz 4:bz 5:Wr 6:br 7:Wh 8:bh 9:W_lin 10:b_lin
  const float* x    = (const float*)d_in[0];
  const float* Wz   = (const float*)d_in[3];
  const float* bz   = (const float*)d_in[4];
  const float* Wh   = (const float*)d_in[7];
  const float* bh   = (const float*)d_in[8];
  const float* wlin = (const float*)d_in[9];
  const float* blin = (const float*)d_in[10];
  float* out = (float*)d_out;

  const int n = in_sizes[0] / 64;          // 200000 nodes

  half_t* wsZ = (half_t*)d_ws;             // 8 frags * 32 lanes * 16 halves = 8KB
  half_t* wsH = wsZ + 8 * 32 * 16;         // +8KB

  prep_weights<<<2, 256, 0, stream>>>(Wz, Wh, wsZ, wsH);

  const int numTiles = (n + 15) / 16;      // 12500
  const int blocks   = (numTiles + 7) / 8; // 8 waves (tiles) per 256-thr block
  dcrnn_fused<<<blocks, 256, 0, stream>>>(x, bz, bh, wlin, blin, wsZ, wsH, out, n);
}
